// EdgeClassifier_52072183497375
// MI455X (gfx1250) — matrix operand order repaired and checked by reading, not compiled, not run on hardware
//
#include <hip/hip_runtime.h>
#include <stdint.h>

#define HD      128
#define TILE_E  128
#define PAD     136          // row pitch in bf16 elems: 272B = 68 DW, 68 % 64 = 4 -> no bank conflicts
#define LN_EPS  1e-5f

typedef __bf16 bf16_t;
typedef bf16_t v16bf __attribute__((ext_vector_type(16)));
typedef bf16_t v2bf  __attribute__((ext_vector_type(2)));
typedef float  v8f   __attribute__((ext_vector_type(8)));
typedef unsigned u32x4 __attribute__((ext_vector_type(4)));

union Frag { v16bf v; u32x4 q[2]; };

// ---- f32 -> bf16 (RNE), native packed cvt when the builtin exists ----
#if __has_builtin(__builtin_amdgcn_cvt_pk_bf16_f32)
__device__ __forceinline__ unsigned pack_bf16(float lo, float hi) {
  union { v2bf v; unsigned u; } c;
  c.v = __builtin_amdgcn_cvt_pk_bf16_f32(lo, hi);
  return c.u;
}
__device__ __forceinline__ unsigned short f2bf(float f) {
  return (unsigned short)(pack_bf16(f, 0.f) & 0xFFFFu);
}
#else
__device__ __forceinline__ unsigned short f2bf(float f) {
  union { float f; unsigned u; } c; c.f = f;
  unsigned u = c.u;
  u += 0x7FFFu + ((u >> 16) & 1u);
  return (unsigned short)(u >> 16);
}
__device__ __forceinline__ unsigned pack_bf16(float lo, float hi) {
  return (unsigned)f2bf(lo) | ((unsigned)f2bf(hi) << 16);
}
#endif

__device__ __forceinline__ float bf2f(unsigned short h) {
  union { unsigned u; float f; } c; c.u = ((unsigned)h) << 16;
  return c.f;
}

__device__ __forceinline__ v8f wmma_bf16(const Frag& a, const u32x4& q0, const u32x4& q1, v8f c) {
  Frag b; b.q[0] = q0; b.q[1] = q1;
  return __builtin_amdgcn_wmma_f32_16x16x32_bf16(false, a.v, false, b.v, (short)0, c,
                                                 false, false);
}

// C[16x128 strip] += A[16 x 128] * B[128 x 128].
// A rows at sIn[(mbase+m)*PAD + k]; B transposed: sWt[n*PAD + k].
// All 18 ds_load_b128 for a K-step issue back-to-back from inline asm; a
// partial s_wait_dscnt 0x8 releases the first 4 WMMAs while the remaining 8
// loads stay in flight, drained by s_wait_dscnt 0x0 before the last 4 WMMAs.
// B tile stride in LDS = 16*PAD*2 = 4352 B (literal offsets below).
__device__ __forceinline__ void wmma_block(const unsigned short* sIn,
                                           const unsigned short* sWt,
                                           int mbase, int lane, v8f acc[8]) {
  const int l15 = lane & 15;
  const int kha = (lane >> 4) * 16;   // A frag: lanes 16-31 hold K=16..31
  const int khb = (lane >> 4) * 16;   // B frag: lanes 16-31 hold K=16..31
  unsigned pa = (unsigned)(unsigned long long)(const void*)(sIn + (mbase + l15) * PAD + kha);
  unsigned pb = (unsigned)(unsigned long long)(const void*)(sWt + l15 * PAD + khb);
#pragma unroll
  for (int k0 = 0; k0 < HD; k0 += 32) {
    u32x4 a0, a1, b0, b1, b2, b3, b4, b5, b6, b7, b8, b9, b10, b11, b12, b13, b14, b15;
    asm volatile(
        "s_nop 3\n\t"                              // WMMA->ds_load WAR spacing
        "ds_load_b128 %0, %18\n\t"
        "ds_load_b128 %1, %18 offset:16\n\t"
        "ds_load_b128 %2, %19\n\t"
        "ds_load_b128 %3, %19 offset:16\n\t"
        "ds_load_b128 %4, %19 offset:4352\n\t"
        "ds_load_b128 %5, %19 offset:4368\n\t"
        "ds_load_b128 %6, %19 offset:8704\n\t"
        "ds_load_b128 %7, %19 offset:8720\n\t"
        "ds_load_b128 %8, %19 offset:13056\n\t"
        "ds_load_b128 %9, %19 offset:13072\n\t"
        "ds_load_b128 %10, %19 offset:17408\n\t"
        "ds_load_b128 %11, %19 offset:17424\n\t"
        "ds_load_b128 %12, %19 offset:21760\n\t"
        "ds_load_b128 %13, %19 offset:21776\n\t"
        "ds_load_b128 %14, %19 offset:26112\n\t"
        "ds_load_b128 %15, %19 offset:26128\n\t"
        "ds_load_b128 %16, %19 offset:30464\n\t"
        "ds_load_b128 %17, %19 offset:30480\n\t"
        "s_wait_dscnt 0x8"                         // A + B0..B3 landed; 8 loads in flight
        : "=v"(a0), "=v"(a1), "=v"(b0), "=v"(b1), "=v"(b2), "=v"(b3), "=v"(b4), "=v"(b5),
          "=v"(b6), "=v"(b7), "=v"(b8), "=v"(b9), "=v"(b10), "=v"(b11), "=v"(b12),
          "=v"(b13), "=v"(b14), "=v"(b15)
        : "v"(pa), "v"(pb)
        : "memory");
    Frag fa; fa.q[0] = a0; fa.q[1] = a1;
    acc[0] = wmma_bf16(fa, b0, b1, acc[0]);
    acc[1] = wmma_bf16(fa, b2, b3, acc[1]);
    acc[2] = wmma_bf16(fa, b4, b5, acc[2]);
    acc[3] = wmma_bf16(fa, b6, b7, acc[3]);
    asm volatile("s_wait_dscnt 0x0"                // drain B4..B7; ties the late frags
        : "+v"(b8), "+v"(b9), "+v"(b10), "+v"(b11),
          "+v"(b12), "+v"(b13), "+v"(b14), "+v"(b15));
    acc[4] = wmma_bf16(fa, b8, b9, acc[4]);
    acc[5] = wmma_bf16(fa, b10, b11, acc[5]);
    acc[6] = wmma_bf16(fa, b12, b13, acc[6]);
    acc[7] = wmma_bf16(fa, b14, b15, acc[7]);
    pa += 64;                                      // 32 bf16 elems
    pb += 64;
  }
}

// bias add + layernorm + relu over the 16 rows held by this wave; store bf16.
__device__ __forceinline__ void ln_relu_store(v8f acc[8],
                                              const float* __restrict__ bias,
                                              const float* __restrict__ g,
                                              const float* __restrict__ be,
                                              unsigned short* dst, int mbase, int lane) {
  const int l15 = lane & 15;
  const int mh  = (lane >> 4) * 8;    // C/D layout: lanes 16-31 hold rows M=8..15
#pragma unroll
  for (int t = 0; t < 8; ++t) {
    float bt = bias[t * 16 + l15];
#pragma unroll
    for (int r = 0; r < 8; ++r) acc[t][r] += bt;
  }
  float mean[8], scl[8];
#pragma unroll
  for (int r = 0; r < 8; ++r) {
    float s = 0.f;
#pragma unroll
    for (int t = 0; t < 8; ++t) s += acc[t][r];
    s += __shfl_xor(s, 1, 32);
    s += __shfl_xor(s, 2, 32);
    s += __shfl_xor(s, 4, 32);
    s += __shfl_xor(s, 8, 32);        // stays within 16-lane half
    mean[r] = s * (1.0f / 128.0f);
  }
#pragma unroll
  for (int r = 0; r < 8; ++r) {
    float q = 0.f;
#pragma unroll
    for (int t = 0; t < 8; ++t) { float d = acc[t][r] - mean[r]; q += d * d; }
    q += __shfl_xor(q, 1, 32);
    q += __shfl_xor(q, 2, 32);
    q += __shfl_xor(q, 4, 32);
    q += __shfl_xor(q, 8, 32);
    scl[r] = rsqrtf(q * (1.0f / 128.0f) + LN_EPS);
  }
#pragma unroll
  for (int t = 0; t < 8; ++t) {
    float gt = g[t * 16 + l15];
    float bt = be[t * 16 + l15];
#pragma unroll
    for (int r = 0; r < 8; ++r) {
      float y = (acc[t][r] - mean[r]) * scl[r] * gt + bt;
      y = y > 0.f ? y : 0.f;
      dst[(mbase + mh + r) * PAD + t * 16 + l15] = f2bf(y);
    }
  }
}

// transpose-stage a 128x128 f32 weight block into LDS as bf16 [n][k]
__device__ __forceinline__ void stage_w(const float* __restrict__ Wg,
                                        unsigned short* sWt, int tid) {
  for (int i = tid * 4; i < HD * HD; i += 1024) {   // Wg row-major [k][n]
    int kk = i >> 7, n = i & 127;                   // n multiple of 4
    float4 w = *(const float4*)(Wg + i);
    sWt[(n + 0) * PAD + kk] = f2bf(w.x);
    sWt[(n + 1) * PAD + kk] = f2bf(w.y);
    sWt[(n + 2) * PAD + kk] = f2bf(w.z);
    sWt[(n + 3) * PAD + kk] = f2bf(w.w);
  }
}

__global__ __launch_bounds__(256)
__attribute__((amdgpu_waves_per_eu(1, 4)))
void edge_mlp_wmma_kernel(const float* __restrict__ x,
                          const long long* __restrict__ ei,
                          const float* __restrict__ edge,
                          const float* __restrict__ W0, const float* __restrict__ b0,
                          const float* __restrict__ g0, const float* __restrict__ be0,
                          const float* __restrict__ W1, const float* __restrict__ b1,
                          const float* __restrict__ g1, const float* __restrict__ be1,
                          const float* __restrict__ W2, const float* __restrict__ b2,
                          const float* __restrict__ g2, const float* __restrict__ be2,
                          const float* __restrict__ W3, const float* __restrict__ b3,
                          float* __restrict__ out, int E) {
  __shared__ unsigned short sA [TILE_E * PAD];
  __shared__ unsigned short sW [TILE_E * PAD];
  __shared__ unsigned short sH0[TILE_E * PAD];
  __shared__ unsigned short sH1[TILE_E * PAD];
  __shared__ float sw3[HD];

  const int tid   = threadIdx.x;
  const int lane  = tid & 31;
  const int wave  = tid >> 5;
  const int mbase = wave * 16;
  const int ebase = blockIdx.x * TILE_E;

  if (tid < HD) sw3[tid] = W3[tid];

  v8f acc[8];
#pragma unroll
  for (int t = 0; t < 8; ++t)
#pragma unroll
    for (int r = 0; r < 8; ++r) acc[t][r] = 0.f;

  // ---- layer 0: [x[src] | x[dst] | edge] @ W0, as 3 accumulated 128-K chunks ----
  const int e_local = tid >> 1;
  const int half    = tid & 1;
  const int e       = ebase + e_local;
  long long si = 0, di = 0;
  if (e < E) { si = ei[e]; di = ei[(long long)E + e]; }

  for (int c = 0; c < 3; ++c) {
    const float* rowp;
    if (c == 0)      rowp = x + (size_t)si * HD;
    else if (c == 1) rowp = x + (size_t)di * HD;
    else             rowp = edge + (size_t)e * HD;

    unsigned short* da = sA + e_local * PAD + half * 64;
    const float*    sp = rowp + half * 64;
    if (e < E) {
#pragma unroll
      for (int i = 0; i < 16; ++i) {
        float4 v = *(const float4*)(sp + i * 4);
        uint2 u;
        u.x = pack_bf16(v.x, v.y);
        u.y = pack_bf16(v.z, v.w);
        *(uint2*)(da + i * 4) = u;
      }
    } else {
      uint2 z; z.x = 0u; z.y = 0u;
#pragma unroll
      for (int i = 0; i < 16; ++i) *(uint2*)(da + i * 4) = z;
    }
    // stage W0 chunk (rows c*128..c*128+127), transposed, bf16
    for (int i = tid * 4; i < HD * HD; i += 1024) {
      int kk = i >> 7, n = i & 127;
      float4 w = *(const float4*)(W0 + c * HD * HD + i);
      sW[(n + 0) * PAD + kk] = f2bf(w.x);
      sW[(n + 1) * PAD + kk] = f2bf(w.y);
      sW[(n + 2) * PAD + kk] = f2bf(w.z);
      sW[(n + 3) * PAD + kk] = f2bf(w.w);
    }
    // prefetch next chunk's gather rows so HBM latency overlaps the WMMAs
    if (c < 2 && e < E) {
      const float* nxt = (c == 0) ? (x + (size_t)di * HD) : (edge + (size_t)e * HD);
      const char* pp = (const char*)(nxt + half * 64);
      __builtin_prefetch(pp, 0, 0);
      __builtin_prefetch(pp + 128, 0, 0);
    }
    __syncthreads();
    wmma_block(sA, sW, mbase, lane, acc);
    __syncthreads();
  }

  ln_relu_store(acc, b0, g0, be0, sH0, mbase, lane);
  stage_w(W1, sW, tid);
  __syncthreads();

  // ---- layer 1 ----
#pragma unroll
  for (int t = 0; t < 8; ++t)
#pragma unroll
    for (int r = 0; r < 8; ++r) acc[t][r] = 0.f;
  wmma_block(sH0, sW, mbase, lane, acc);
  __syncthreads();
  ln_relu_store(acc, b1, g1, be1, sH1, mbase, lane);
  stage_w(W2, sW, tid);
  __syncthreads();

  // ---- layer 2 ----
#pragma unroll
  for (int t = 0; t < 8; ++t)
#pragma unroll
    for (int r = 0; r < 8; ++r) acc[t][r] = 0.f;
  wmma_block(sH1, sW, mbase, lane, acc);
  __syncthreads();
  ln_relu_store(acc, b2, g2, be2, sA, mbase, lane);   // sA free after layer 0
  __syncthreads();

  // ---- final projection: out[e] = h2 . w3 + b3 ----
  if (tid < TILE_E) {
    int eo = ebase + tid;
    if (eo < E) {
      float s = b3[0];
#pragma unroll
      for (int i = 0; i < 16; ++i) {
        uint4 q = *(const uint4*)(sA + tid * PAD + i * 8);
        const float* wp = sw3 + i * 8;
        unsigned a[4] = { q.x, q.y, q.z, q.w };
#pragma unroll
        for (int j = 0; j < 4; ++j) {
          s += bf2f((unsigned short)(a[j] & 0xFFFFu)) * wp[j * 2];
          s += bf2f((unsigned short)(a[j] >> 16))     * wp[j * 2 + 1];
        }
      }
      out[eo] = s;
    }
  }
}

extern "C" void kernel_launch(void* const* d_in, const int* in_sizes, int n_in,
                              void* d_out, int out_size, void* d_ws, size_t ws_size,
                              hipStream_t stream) {
  (void)n_in; (void)out_size; (void)d_ws; (void)ws_size;
  const float*     x    = (const float*)d_in[0];
  const long long* ei   = (const long long*)d_in[1];
  const float*     edge = (const float*)d_in[2];
  const float* W0 = (const float*)d_in[3];  const float* b0  = (const float*)d_in[4];
  const float* g0 = (const float*)d_in[5];  const float* be0 = (const float*)d_in[6];
  const float* W1 = (const float*)d_in[7];  const float* b1  = (const float*)d_in[8];
  const float* g1 = (const float*)d_in[9];  const float* be1 = (const float*)d_in[10];
  const float* W2 = (const float*)d_in[11]; const float* b2  = (const float*)d_in[12];
  const float* g2 = (const float*)d_in[13]; const float* be2 = (const float*)d_in[14];
  const float* W3 = (const float*)d_in[15]; const float* b3  = (const float*)d_in[16];
  float* out = (float*)d_out;

  const int E = in_sizes[1] / 2;           // edge_index is [2, E]
  const int grid = (E + TILE_E - 1) / TILE_E;
  hipLaunchKernelGGL(edge_mlp_wmma_kernel, dim3(grid), dim3(256), 0, stream,
                     x, ei, edge,
                     W0, b0, g0, be0,
                     W1, b1, g1, be1,
                     W2, b2, g2, be2,
                     W3, b3, out, E);
}
